// PMS_1_25898652795298
// MI455X (gfx1250) — matrix operand order repaired and checked by reading, not compiled, not run on hardware
//
#include <hip/hip_runtime.h>

// ---------------------------------------------------------------------------
// PMS block for MI455X (gfx1250): implicit-GEMM convs on V_WMMA_F32_16X16X32_BF16
// with ASYNCcnt-tracked global->LDS staging when the toolchain exposes it.
//
// Pipeline (all on `stream`):
//   nchw->nhwc  ->  9x [ pack_act(cond) -> conv3 -> conv3(+residual) ]
//   with MFE after resblocks {1,3,5,7,8}:
//      stats(max/avg) -> gate(SE+sigmoid+affine)
//      1x1,1x1 | 3x3,3x3 | 5x5,5x5  (branch sums accumulated into S)
//      M = gate * S
//   final: out = M*map + x   (NHWC -> NCHW)
// ---------------------------------------------------------------------------

typedef __attribute__((ext_vector_type(16))) __bf16 v16bf;
typedef __attribute__((ext_vector_type(8)))  float  v8f;
typedef __attribute__((ext_vector_type(4)))  unsigned u32x4;
typedef __attribute__((ext_vector_type(4)))  int      v4i;

union FragU { v16bf v; u32x4 q[2]; };

#define ACT_NONE  0
#define ACT_RELU  1
#define ACT_LRELU 2

constexpr int BN  = 4;     // batch
constexpr int CCH = 64;    // channels
constexpr int HH  = 128;
constexpr int WW  = 128;
constexpr int HWN = HH * WW;                  // 16384 = 2^14
constexpr int NTOT = BN * CCH * HWN;          // 4,194,304 = 2^22
constexpr float MODSCALE = 0.041666666666666664f;  // 1/sqrt(64*9)

// Async global->LDS path (GLOBAL_LOAD_ASYNC_TO_LDS_B128 + s_wait_asynccnt),
// guarded so the build falls back to load/ds_store staging if unavailable.
#if __has_builtin(__builtin_amdgcn_global_load_async_to_lds_b128) && \
    __has_builtin(__builtin_amdgcn_s_wait_asynccnt)
#define USE_ASYNC_LDS 1
#else
#define USE_ASYNC_LDS 0
#endif

__device__ __forceinline__ unsigned short f2bf(float x) {
  union { float f; unsigned u; } a; a.f = x;
  unsigned r = a.u + 0x7FFFu + ((a.u >> 16) & 1u);   // round-to-nearest-even
  return (unsigned short)(r >> 16);
}

__device__ __forceinline__ void cp16(unsigned short* ldst,
                                     const unsigned short* gsrc) {
#if USE_ASYNC_LDS
  __builtin_amdgcn_global_load_async_to_lds_b128(
      (__attribute__((address_space(1))) v4i*)gsrc,
      (__attribute__((address_space(3))) v4i*)ldst, 0, 0);
#else
  *(u32x4*)ldst = *(const u32x4*)gsrc;
#endif
}

__device__ __forceinline__ void stage_fence() {
#if USE_ASYNC_LDS
  __builtin_amdgcn_s_wait_asynccnt(0);
#endif
  __syncthreads();
}

// ---------------------------------------------------------------------------
// Implicit-GEMM conv, C=64 -> C=64, kernel KSxKS, pad KS/2, stride 1.
// Tile per workgroup: 8 rows x 16 cols of output pixels, all 64 out channels.
// Wave w computes row w: M=16 pixels, 4 WMMA N-tiles, K = KS*KS*64 in 32-chunks.
// ---------------------------------------------------------------------------
template <int KS, bool WLDS>
__global__ __launch_bounds__(256) void conv_wmma_kernel(
    const unsigned short* __restrict__ in,    // bf16 NHWC [B][H][W][64]
    const unsigned short* __restrict__ wpk,   // packed B fragments (bf16)
    const float* __restrict__ bias,           // [64]
    const float* __restrict__ addsrc,         // f32 NHWC or null
    float* __restrict__ outf,                 // f32 NHWC or null
    unsigned short* __restrict__ outbf,       // bf16 NHWC or null
    const float* __restrict__ cond,           // [B*64] scale for outbf, or null
    int act)
{
  constexpr int KR = KS / 2;
  constexpr int TH = 8, TW = 16;
  constexpr int IH = TH + 2 * KR, IW = TW + 2 * KR;
  constexpr int NK = KS * KS * 2;             // K-steps of 32 (always even)

  extern __shared__ unsigned short smem[];
  unsigned short* sIn = smem;                 // IH*IW*64 bf16
  unsigned short* sW  = smem + IH * IW * 64;  // KS*KS*4096 bf16 (if WLDS)

  const int tid = threadIdx.x;
  const int b   = blockIdx.z;
  const int h0  = blockIdx.y * TH;
  const int w0  = blockIdx.x * TW;

  // ---- stage activation tile (+halo, zero-padded) into LDS, 16B chunks ----
  {
    const int nchunk = IH * IW * 8;           // 8 bf16 per chunk
    for (int ch = tid; ch < nchunk; ch += 256) {
      const int pix = ch >> 3, c8 = ch & 7;
      const int ihh = pix / IW, iww = pix % IW;
      const int h = h0 + ihh - KR, w = w0 + iww - KR;
      if (h >= 0 && h < HH && w >= 0 && w < WW) {
        cp16(&sIn[ch << 3],
             &in[(((b * HH + h) * WW + w) << 6) + (c8 << 3)]);
      } else {
        u32x4 z = {0u, 0u, 0u, 0u};
        *(u32x4*)&sIn[ch << 3] = z;           // padding: direct LDS store
      }
    }
  }
  if constexpr (WLDS) {                       // stage packed weights (linear copy)
    const int nchunk = KS * KS * 512;
    for (int ch = tid; ch < nchunk; ch += 256)
      cp16(&sW[ch << 3], &wpk[ch << 3]);
  }
  stage_fence();

  const int wave = tid >> 5;
  const int lane = tid & 31;
  const int m    = lane & 15;                 // A-row (pixel) for this lane
  const int koff = (lane >> 4) << 3;          // K sub-offset per ISA A layout
  const int mro  = (lane >> 4) << 3;          // C/D row offset
  const int r    = wave;                      // output row within tile

  v8f acc0 = {0,0,0,0,0,0,0,0};
  v8f acc1 = acc0, acc2 = acc0, acc3 = acc0;

  #pragma unroll 2
  for (int kstep = 0; kstep < NK; ++kstep) {
    const int tap = kstep >> 1, chalf = kstep & 1;
    const int ky = tap / KS, kx = tap % KS;

    // A fragment (16x32 bf16): lanes 0-15 -> K {0..7,16..23}; 16-31 -> {8..15,24..31}
    FragU A;
    const unsigned short* pa =
        &sIn[(((r + ky) * IW) + (m + kx)) * 64 + (chalf << 5) + koff];
    A.q[0] = *(const u32x4*)(pa);
    A.q[1] = *(const u32x4*)(pa + 16);

    // B fragments: frag f = kstep*4 + nt; lane-major 32B per lane
    FragU B0, B1, B2, B3;
    const int fo = (kstep << 2) * 512 + lane * 16;
    if constexpr (WLDS) {
      const unsigned short* pb = &sW[fo];
      B0.q[0] = *(const u32x4*)(pb);        B0.q[1] = *(const u32x4*)(pb + 8);
      B1.q[0] = *(const u32x4*)(pb + 512);  B1.q[1] = *(const u32x4*)(pb + 520);
      B2.q[0] = *(const u32x4*)(pb + 1024); B2.q[1] = *(const u32x4*)(pb + 1032);
      B3.q[0] = *(const u32x4*)(pb + 1536); B3.q[1] = *(const u32x4*)(pb + 1544);
    } else {
      const unsigned short* pb = &wpk[fo];
      if (kstep + 1 < NK)                    // keep the B-stream ahead in cache
        __builtin_prefetch(pb + 2048, 0, 3);
      B0.q[0] = *(const u32x4*)(pb);        B0.q[1] = *(const u32x4*)(pb + 8);
      B1.q[0] = *(const u32x4*)(pb + 512);  B1.q[1] = *(const u32x4*)(pb + 520);
      B2.q[0] = *(const u32x4*)(pb + 1024); B2.q[1] = *(const u32x4*)(pb + 1032);
      B3.q[0] = *(const u32x4*)(pb + 1536); B3.q[1] = *(const u32x4*)(pb + 1544);
    }

    acc0 = __builtin_amdgcn_wmma_f32_16x16x32_bf16(false, A.v, false, B0.v, (short)0, acc0, false, false);
    acc1 = __builtin_amdgcn_wmma_f32_16x16x32_bf16(false, A.v, false, B1.v, (short)0, acc1, false, false);
    acc2 = __builtin_amdgcn_wmma_f32_16x16x32_bf16(false, A.v, false, B2.v, (short)0, acc2, false, false);
    acc3 = __builtin_amdgcn_wmma_f32_16x16x32_bf16(false, A.v, false, B3.v, (short)0, acc3, false, false);
  }

  // ---- epilogue: bias + activation + (residual/branch add) + stores ----
  const int h = h0 + r;
  v8f accs[4] = {acc0, acc1, acc2, acc3};
  #pragma unroll
  for (int nt = 0; nt < 4; ++nt) {
    const int n = (nt << 4) + (lane & 15);
    const float bn = bias[n];
    const float cs = cond ? cond[(b << 6) + n] : 1.0f;
    #pragma unroll
    for (int v = 0; v < 8; ++v) {
      const int mm  = v + mro;
      const int w   = w0 + mm;
      const int idx = (((b * HH + h) * WW + w) << 6) + n;
      float val = accs[nt][v] + bn;
      if (act == ACT_RELU)       val = fmaxf(val, 0.0f);
      else if (act == ACT_LRELU) val = (val > 0.0f) ? val : 0.01f * val;
      if (addsrc) val += addsrc[idx];
      if (outf)   outf[idx] = val;
      if (outbf)  outbf[idx] = f2bf(val * cs);
    }
  }
}

// ---- pack raw OIHW f32 weights into WMMA B-fragment order (bf16) -----------
__global__ void pack_w_kernel(const float* __restrict__ w,
                              unsigned short* __restrict__ wpk,
                              int KS, float scale, int n) {
  int i = blockIdx.x * 256 + threadIdx.x;
  if (i >= n) return;
  const int f = i >> 9, e = i & 511;
  const int lane = e >> 4, t = e & 15;
  const int kstep = f >> 2, nt = f & 3;
  const int tap = kstep >> 1, chalf = kstep & 1;
  const int ky = tap / KS, kx = tap % KS;
  const int nn = (nt << 4) + (lane & 15);
  const int c  = (chalf << 5) + ((lane >> 4) << 3) + ((t >> 3) << 4) + (t & 7);
  wpk[i] = f2bf(w[((nn << 6) + c) * KS * KS + ky * KS + kx] * scale);
}

// ---- pack f32 NHWC master -> bf16, with optional per-(b,c) cond scale ------
__global__ void pack_act_kernel(const float* __restrict__ src,
                                const float* __restrict__ cond,
                                unsigned short* __restrict__ dst, int n) {
  int i = blockIdx.x * 256 + threadIdx.x;
  if (i >= n) return;
  const int c = i & 63, b = i >> 20;
  const float s = cond ? cond[(b << 6) + c] : 1.0f;
  dst[i] = f2bf(src[i] * s);
}

__global__ void nchw2nhwc_kernel(const float* __restrict__ src,
                                 float* __restrict__ dst, int n) {
  int i = blockIdx.x * 256 + threadIdx.x;
  if (i >= n) return;
  const int b = i >> 20, c = (i >> 14) & 63, p = i & 16383;
  dst[(b << 20) + (p << 6) + c] = src[i];
}

__global__ void final_kernel(const float* __restrict__ M,
                             const float* __restrict__ mp,
                             const float* __restrict__ x,
                             float* __restrict__ out, int n) {
  int i = blockIdx.x * 256 + threadIdx.x;
  if (i >= n) return;
  const int b = i >> 20, c = (i >> 14) & 63, p = i & 16383;
  out[i] = M[(b << 20) + (p << 6) + c] * mp[(b << 14) + p] + x[i];
}

// ---- channel stats: max & mean over HW per (b,c) --------------------------
__global__ __launch_bounds__(256) void stats_kernel(
    const float* __restrict__ M, float* __restrict__ mx, float* __restrict__ av) {
  const int bc = blockIdx.x, b = bc >> 6, c = bc & 63;
  const float* base = M + (b << 20) + c;
  float vmax = -3.4e38f, vsum = 0.0f;
  for (int p = threadIdx.x; p < HWN; p += 256) {
    const float v = base[p << 6];
    vmax = fmaxf(vmax, v);
    vsum += v;
  }
  __shared__ float smax[256], ssum[256];
  smax[threadIdx.x] = vmax; ssum[threadIdx.x] = vsum;
  __syncthreads();
  for (int s = 128; s > 0; s >>= 1) {
    if (threadIdx.x < s) {
      smax[threadIdx.x] = fmaxf(smax[threadIdx.x], smax[threadIdx.x + s]);
      ssum[threadIdx.x] += ssum[threadIdx.x + s];
    }
    __syncthreads();
  }
  if (threadIdx.x == 0) { mx[bc] = smax[0]; av[bc] = ssum[0] * (1.0f / HWN); }
}

// ---- SE gate: sigmoid(se(mx)+se(av)) through the FC1 affine ---------------
__global__ void gate_kernel(const float* __restrict__ mx,
                            const float* __restrict__ av,
                            const float* __restrict__ cw1,  // [4][64]
                            const float* __restrict__ cw2,  // [64][4]
                            const float* __restrict__ fw,   // [2]
                            const float* __restrict__ fb,   // [2]
                            float* __restrict__ gate) {
  const int t = threadIdx.x;            // 256 = B*C
  const int b = t >> 6, o = t & 63;
  float sm = 0.0f, sa = 0.0f;
  #pragma unroll
  for (int hh = 0; hh < 4; ++hh) {
    float rm = 0.0f, ra = 0.0f;
    for (int i = 0; i < 64; ++i) {
      const float w = cw1[hh * 64 + i];
      rm += w * mx[(b << 6) + i];
      ra += w * av[(b << 6) + i];
    }
    const float w2 = cw2[o * 4 + hh];
    sm += w2 * fmaxf(rm, 0.0f);
    sa += w2 * fmaxf(ra, 0.0f);
  }
  const float xll = 1.0f / (1.0f + __expf(-(sm + sa)));
  gate[t] = fw[1] * (fw[0] * xll + fb[0]) + fb[1];
}

__global__ void combine_kernel(const float* __restrict__ S,
                               const float* __restrict__ gate,
                               float* __restrict__ M, int n) {
  int i = blockIdx.x * 256 + threadIdx.x;
  if (i >= n) return;
  const int c = i & 63, b = i >> 20;
  M[i] = gate[(b << 6) + c] * S[i];
}

// ---------------------------------------------------------------------------
// Orchestration
// ---------------------------------------------------------------------------
extern "C" void kernel_launch(void* const* d_in, const int* in_sizes, int n_in,
                              void* d_out, int out_size, void* d_ws, size_t ws_size,
                              hipStream_t stream) {
  const float* x    = (const float*)d_in[0];
  const float* cf1  = (const float*)d_in[1];
  const float* cf2  = (const float*)d_in[2];
  const float* mp   = (const float*)d_in[3];
  const float* resw = (const float*)d_in[4];
  const float* resb = (const float*)d_in[5];
  const float* mw1  = (const float*)d_in[6];
  const float* mb1  = (const float*)d_in[7];
  const float* mw3  = (const float*)d_in[8];
  const float* mb3  = (const float*)d_in[9];
  const float* mw5  = (const float*)d_in[10];
  const float* mb5  = (const float*)d_in[11];
  const float* caw1 = (const float*)d_in[12];
  const float* caw2 = (const float*)d_in[13];
  const float* fcw  = (const float*)d_in[14];
  const float* fcb  = (const float*)d_in[15];
  float* out = (float*)d_out;

  // workspace layout (~56.3 MB)
  char* ws = (char*)d_ws;
  constexpr size_t SZ_F32 = (size_t)NTOT * 4;   // 16 MB
  constexpr size_t SZ_BF  = (size_t)NTOT * 2;   //  8 MB
  float*          M    = (float*)(ws);
  float*          S    = (float*)(ws + SZ_F32);
  unsigned short* Mbf  = (unsigned short*)(ws + 2 * SZ_F32);
  unsigned short* bufA = (unsigned short*)(ws + 2 * SZ_F32 + SZ_BF);
  unsigned short* bufB = (unsigned short*)(ws + 2 * SZ_F32 + 2 * SZ_BF);
  unsigned short* Wpk  = (unsigned short*)(ws + 2 * SZ_F32 + 3 * SZ_BF);
  float* mxb   = (float*)(ws + 2 * SZ_F32 + 3 * SZ_BF + 256 * 1024);
  float* avb   = mxb + 1024;
  float* gateb = avb + 1024;

  const dim3 ewB((NTOT + 255) / 256), ewT(256);
  const dim3 cg(WW / 16, HH / 8, BN), cb(256);
  const size_t sh1 = (size_t)(8  * 16 * 64 +  1 * 4096) * 2;  // 24.5 KB
  const size_t sh3 = (size_t)(10 * 18 * 64 +  9 * 4096) * 2;  // 96.7 KB
  const size_t sh5 = (size_t)(12 * 20 * 64) * 2;              // 30.7 KB

  nchw2nhwc_kernel<<<ewB, ewT, 0, stream>>>(x, M, NTOT);

  const int mfe_after[9] = {-1, 0, -1, 1, -1, 2, -1, 3, 4};
  for (int i = 0; i < 9; ++i) {
    const float* cond = (i < 5) ? (cf1 + i * 512) : (cf2 + (i - 5) * 512);

    // resblock: r = relu(modconv(M, w0)); M += modconv(r, w1)
    pack_act_kernel<<<ewB, ewT, 0, stream>>>(M, cond, bufA, NTOT);
    {
      const int wn = 9 * 4096;
      pack_w_kernel<<<(wn + 255) / 256, 256, 0, stream>>>(
          resw + (size_t)(i * 2 + 0) * 36864, Wpk, 3, MODSCALE, wn);
      conv_wmma_kernel<3, true><<<cg, cb, sh3, stream>>>(
          bufA, Wpk, resb + (i * 2 + 0) * 64, nullptr, nullptr, bufB,
          cond + 256, ACT_RELU);
      pack_w_kernel<<<(wn + 255) / 256, 256, 0, stream>>>(
          resw + (size_t)(i * 2 + 1) * 36864, Wpk, 3, MODSCALE, wn);
      conv_wmma_kernel<3, true><<<cg, cb, sh3, stream>>>(
          bufB, Wpk, resb + (i * 2 + 1) * 64, M, M, Mbf, nullptr, ACT_NONE);
    }

    const int j = mfe_after[i];
    if (j >= 0) {
      stats_kernel<<<256, 256, 0, stream>>>(M, mxb, avb);
      gate_kernel<<<1, 256, 0, stream>>>(mxb, avb, caw1 + j * 256,
                                         caw2 + j * 256, fcw + j * 2,
                                         fcb + j * 2, gateb);
      { // 1x1 branch -> S (write)
        const int wn = 1 * 4096;
        pack_w_kernel<<<(wn + 255) / 256, 256, 0, stream>>>(
            mw1 + (size_t)(j * 2 + 0) * 4096, Wpk, 1, 1.0f, wn);
        conv_wmma_kernel<1, true><<<cg, cb, sh1, stream>>>(
            Mbf, Wpk, mb1 + (j * 2 + 0) * 64, nullptr, nullptr, bufA, nullptr, ACT_LRELU);
        pack_w_kernel<<<(wn + 255) / 256, 256, 0, stream>>>(
            mw1 + (size_t)(j * 2 + 1) * 4096, Wpk, 1, 1.0f, wn);
        conv_wmma_kernel<1, true><<<cg, cb, sh1, stream>>>(
            bufA, Wpk, mb1 + (j * 2 + 1) * 64, nullptr, S, nullptr, nullptr, ACT_NONE);
      }
      { // 3x3 branch -> S (accumulate)
        const int wn = 9 * 4096;
        pack_w_kernel<<<(wn + 255) / 256, 256, 0, stream>>>(
            mw3 + (size_t)(j * 2 + 0) * 36864, Wpk, 3, 1.0f, wn);
        conv_wmma_kernel<3, true><<<cg, cb, sh3, stream>>>(
            Mbf, Wpk, mb3 + (j * 2 + 0) * 64, nullptr, nullptr, bufA, nullptr, ACT_LRELU);
        pack_w_kernel<<<(wn + 255) / 256, 256, 0, stream>>>(
            mw3 + (size_t)(j * 2 + 1) * 36864, Wpk, 3, 1.0f, wn);
        conv_wmma_kernel<3, true><<<cg, cb, sh3, stream>>>(
            bufA, Wpk, mb3 + (j * 2 + 1) * 64, S, S, nullptr, nullptr, ACT_NONE);
      }
      { // 5x5 branch -> S (accumulate); weights streamed from L2 (200 KB)
        const int wn = 25 * 4096;
        pack_w_kernel<<<(wn + 255) / 256, 256, 0, stream>>>(
            mw5 + (size_t)(j * 2 + 0) * 102400, Wpk, 5, 1.0f, wn);
        conv_wmma_kernel<5, false><<<cg, cb, sh5, stream>>>(
            Mbf, Wpk, mb5 + (j * 2 + 0) * 64, nullptr, nullptr, bufA, nullptr, ACT_LRELU);
        pack_w_kernel<<<(wn + 255) / 256, 256, 0, stream>>>(
            mw5 + (size_t)(j * 2 + 1) * 102400, Wpk, 5, 1.0f, wn);
        conv_wmma_kernel<5, false><<<cg, cb, sh5, stream>>>(
            bufA, Wpk, mb5 + (j * 2 + 1) * 64, S, S, nullptr, nullptr, ACT_NONE);
      }
      combine_kernel<<<ewB, ewT, 0, stream>>>(S, gateb, M, NTOT);
    }
  }

  final_kernel<<<ewB, ewT, 0, stream>>>(M, mp, x, out, NTOT);
}
